// Network_80049600463282
// MI455X (gfx1250) — matrix operand order repaired and checked by reading, not compiled, not run on hardware
//
#include <hip/hip_runtime.h>

// ---------------------------------------------------------------------------
// LSTM language model forward for MI455X (gfx1250), wave32 + WMMA.
//   tokens[B,T] -> one-hot gather -> LSTM(T=512,H=512) -> bn -> relu(W1)+b1
//   -> W2+b2 -> bn.   B=16, V=4096, H=512, DH=200.
// ---------------------------------------------------------------------------

typedef __attribute__((ext_vector_type(16))) _Float16 v16h;
typedef __attribute__((ext_vector_type(8)))  _Float16 v8h;
typedef __attribute__((ext_vector_type(8)))  float    v8f;

#define VSZ 4096
#define BB  16
#define TT  512
#define HH  512
#define GG  2048          // 4*H gate width
#define DH  200
#define DHP 224           // DH padded to 7 K-chunks of 32
#define BT  (BB*TT)       // 8192
#define HP  520           // padded LDS row (f16) -> 1040B stride, 4-bank rotate
#define BNS 0.99950037f   // 1/sqrt(1+1e-3)

// workspace layout (bytes)
#define WS_WHT  ((size_t)0)                         // f16 [GG][HH]  gate-permuted, K-contig
#define WS_W1T  (WS_WHT + (size_t)GG*HH*2)          // f16 [DHP][HH] K-contig
#define WS_W2T  (WS_W1T + (size_t)DHP*HH*2)         // f16 [VSZ][DHP] K-contig
#define WS_HS   (WS_W2T + (size_t)VSZ*DHP*2)        // f16 [BT][HH]  bn-scaled h states
#define WS_H1   (WS_HS  + (size_t)BT*HH*2)          // f16 [BT][DHP] relu hidden
#define WS_BIAS (WS_H1  + (size_t)BT*DHP*2)         // f32 [GG]      permuted bias (+1 on f)

__device__ __forceinline__ float sigm(float x) { return 1.0f / (1.0f + __expf(-x)); }

__device__ __forceinline__ v16h pack16(v8h lo, v8h hi) {
  v16h r;
#pragma unroll
  for (int i = 0; i < 8; ++i) { r[i] = lo[i]; r[i + 8] = hi[i]; }
  return r;
}

// A-fragment (16x32 f16): lanes 0-15 hold M=lane, K = {0..7,16..23}; lanes 16-31 K={8..15,24..31}
__device__ __forceinline__ v16h load_a(const _Float16* rowbase, int kk, int hi) {
  const _Float16* p = rowbase + kk * 32 + hi * 8;
  return pack16(*(const v8h*)p, *(const v8h*)(p + 16));
}
// B-fragment (32x16 f16), B stored K-contiguous per column: lane N=l%16, K = {0..7,16..23} (l<16) / {8..15,24..31}
__device__ __forceinline__ v16h load_b(const _Float16* colbase, int kk, int hi) {
  const _Float16* p = colbase + kk * 32 + hi * 8;
  return pack16(*(const v8h*)p, *(const v8h*)(p + 16));
}

// ---------------------------------------------------------------------------
// Kernel 1: weight conversion / permutation (one-time, bandwidth trivial)
// ---------------------------------------------------------------------------
__global__ void prep_kernel(const float* __restrict__ lk, const float* __restrict__ lb,
                            const float* __restrict__ W1, const float* __restrict__ W2,
                            char* __restrict__ ws) {
  _Float16* WhT = (_Float16*)(ws + WS_WHT);
  _Float16* W1T = (_Float16*)(ws + WS_W1T);
  _Float16* W2T = (_Float16*)(ws + WS_W2T);
  float*  biasP = (float*)(ws + WS_BIAS);
  const int n0 = GG * HH, n1 = DHP * HH, n2 = VSZ * DHP, n3 = GG;
  const int total = n0 + n1 + n2 + n3;
  for (int i = blockIdx.x * blockDim.x + threadIdx.x; i < total;
       i += gridDim.x * blockDim.x) {
    if (i < n0) {
      int np = i / HH, k = i % HH;
      int src = (np & 3) * HH + (np >> 2);            // gate-interleave permutation
      WhT[i] = (_Float16)lk[(size_t)(VSZ + k) * GG + src];
    } else if (i < n0 + n1) {
      int j = i - n0, n = j / HH, k = j % HH;
      W1T[j] = (_Float16)((n < DH) ? W1[(size_t)k * DH + n] : 0.f);
    } else if (i < n0 + n1 + n2) {
      int j = i - n0 - n1, n = j / DHP, k = j % DHP;
      W2T[j] = (_Float16)((k < DH) ? W2[(size_t)k * VSZ + n] : 0.f);
    } else {
      int np = i - n0 - n1 - n2;
      int g = np & 3, hc = np >> 2;
      biasP[np] = lb[g * HH + hc] + ((g == 2) ? 1.0f : 0.0f);  // fold forget bias
    }
  }
}

// ---------------------------------------------------------------------------
// Kernel 2: the recurrence. 1 workgroup = 32 waves on one WGP.
// Each wave owns 4 N-tiles (64 permuted gate cols = 16 h-cols).
// Per step: z-tile = gather(x-part)+bias + WMMA(h,WhT); gates mixed via
// shfl_xor within lane-quads; c in VGPRs; h double-buffered in LDS.
// ---------------------------------------------------------------------------
__global__ void __launch_bounds__(1024)
lstm_kernel(const int* __restrict__ tokens, const float* __restrict__ lk,
            char* __restrict__ ws) {
  __shared__ _Float16 h_sh[2][BB][HP];                 // 33.3 KB

  const _Float16* WhT = (const _Float16*)(ws + WS_WHT);
  const float* biasP  = (const float*)(ws + WS_BIAS);
  _Float16* hs        = (_Float16*)(ws + WS_HS);

  const int tid = threadIdx.x;
  const int wave = tid >> 5, lane = tid & 31;
  const int l16 = lane & 15, hi = (lane >> 4) & 1, k4 = lane & 3;

  for (int i = tid; i < 2 * BB * HP; i += 1024) ((_Float16*)h_sh)[i] = (_Float16)0.f;

  float biasc[4];
  int   srccol[4], hcol[4];
#pragma unroll
  for (int tt = 0; tt < 4; ++tt) {
    const int c = (wave * 4 + tt) * 16 + l16;          // permuted gate column
    hcol[tt] = c >> 2;
    srccol[tt] = (c & 3) * HH + (c >> 2);              // original column for gather
    biasc[tt] = biasP[c];
  }

  float creg[4][8];
#pragma unroll
  for (int tt = 0; tt < 4; ++tt)
#pragma unroll
    for (int r = 0; r < 8; ++r) creg[tt][r] = 0.f;

  __syncthreads();

  for (int t = 0; t < TT; ++t) {
    const int cur = t & 1, nxt = cur ^ 1;
    int tok[8];
#pragma unroll
    for (int r = 0; r < 8; ++r) tok[r] = tokens[(r + 8 * hi) * TT + t];
    if (t + 1 < TT) {                                  // warm next-step gather rows
      const int tn = tokens[(8 * hi) * TT + t + 1];
      __builtin_prefetch(&lk[(size_t)tn * GG + l16 * 128], 0, 1);
    }

#pragma unroll
    for (int tt = 0; tt < 4; ++tt) {
      v8f acc;
#pragma unroll
      for (int r = 0; r < 8; ++r)                      // x-part: one-hot == row gather
        acc[r] = lk[(size_t)tok[r] * GG + srccol[tt]] + biasc[tt];

      const _Float16* bcol = WhT + (size_t)((wave * 4 + tt) * 16 + l16) * HH;
#pragma unroll
      for (int kk = 0; kk < 16; ++kk) {
        v16h a = load_a(&h_sh[cur][l16][0], kk, hi);   // LDS, conflict-free (row pad)
        v16h b = load_b(bcol, kk, hi);                 // L2-resident weights
        acc = __builtin_amdgcn_wmma_f32_16x16x32_f16(false, a, false, b,
                                                     (short)0, acc, false, false);
      }

      // gate mix: lane-quad holds (i,j,f,o) for one h-column
#pragma unroll
      for (int r = 0; r < 8; ++r) {
        const float zv = acc[r];
        const float z1 = __shfl_xor(zv, 1, 32);
        const float z2 = __shfl_xor(zv, 2, 32);
        const float z3 = __shfl_xor(zv, 3, 32);
        float gi, gj, gf, go;
        if      (k4 == 0) { gi = zv; gj = z1; gf = z2; go = z3; }
        else if (k4 == 1) { gi = z1; gj = zv; gf = z3; go = z2; }
        else if (k4 == 2) { gi = z2; gj = z3; gf = zv; go = z1; }
        else              { gi = z3; gj = z2; gf = z1; go = zv; }
        const float cn = creg[tt][r] * sigm(gf) + sigm(gi) * tanhf(gj);
        creg[tt][r] = cn;
        const float hn = tanhf(cn) * sigm(go);
        if (k4 == 0) {
          const int m = r + 8 * hi;
          h_sh[nxt][m][hcol[tt]] = (_Float16)hn;
          hs[((size_t)m * TT + t) * HH + hcol[tt]] = (_Float16)(hn * BNS);  // bn folded
        }
      }
    }
    __syncthreads();
  }
}

// ---------------------------------------------------------------------------
// Kernel 3: h1 = relu(bn(repr) @ W1 + b1)   [8192,512]x[512,224]
// ---------------------------------------------------------------------------
__global__ void __launch_bounds__(256)
gemm1_kernel(const float* __restrict__ b1, char* __restrict__ ws) {
  const _Float16* A   = (const _Float16*)(ws + WS_HS);
  const _Float16* W1T = (const _Float16*)(ws + WS_W1T);
  _Float16* h1        = (_Float16*)(ws + WS_H1);

  const int wave = threadIdx.x >> 5, lane = threadIdx.x & 31;
  const int l16 = lane & 15, hi = (lane >> 4) & 1;
  const int mbase = (blockIdx.x * 8 + wave) * 16;
  const _Float16* arow = A + (size_t)(mbase + l16) * HH;

  for (int nt = 0; nt < 14; ++nt) {
    const int col = nt * 16 + l16;
    const float bias = (col < DH) ? b1[col] : 0.f;
    v8f acc;
#pragma unroll
    for (int r = 0; r < 8; ++r) acc[r] = bias;
    const _Float16* bcol = W1T + (size_t)col * HH;
#pragma unroll
    for (int kk = 0; kk < 16; ++kk) {
      v16h a = load_a(arow, kk, hi);
      v16h b = load_b(bcol, kk, hi);
      acc = __builtin_amdgcn_wmma_f32_16x16x32_f16(false, a, false, b,
                                                   (short)0, acc, false, false);
    }
#pragma unroll
    for (int r = 0; r < 8; ++r) {
      const float v = acc[r] > 0.f ? acc[r] : 0.f;     // relu
      h1[(size_t)(mbase + r + 8 * hi) * DHP + col] = (_Float16)v;
    }
  }
}

// ---------------------------------------------------------------------------
// Kernel 4: out = bn(h1 @ W2 + b2)   [8192,224]x[224,4096] -> f32 out
// ---------------------------------------------------------------------------
__global__ void __launch_bounds__(256)
gemm2_kernel(const float* __restrict__ b2, char* __restrict__ ws,
             float* __restrict__ out) {
  const _Float16* A   = (const _Float16*)(ws + WS_H1);
  const _Float16* W2T = (const _Float16*)(ws + WS_W2T);

  const int wave = threadIdx.x >> 5, lane = threadIdx.x & 31;
  const int l16 = lane & 15, hi = (lane >> 4) & 1;
  const int mbase = (blockIdx.x * 8 + wave) * 16;
  const _Float16* arow = A + (size_t)(mbase + l16) * DHP;

#pragma unroll
  for (int nt = 0; nt < 4; ++nt) {
    const int col = blockIdx.y * 64 + nt * 16 + l16;
    v8f acc;
#pragma unroll
    for (int r = 0; r < 8; ++r) acc[r] = 0.f;
    const _Float16* bcol = W2T + (size_t)col * DHP;
#pragma unroll
    for (int kk = 0; kk < 7; ++kk) {
      v16h a = load_a(arow, kk, hi);
      v16h b = load_b(bcol, kk, hi);
      acc = __builtin_amdgcn_wmma_f32_16x16x32_f16(false, a, false, b,
                                                   (short)0, acc, false, false);
    }
    const float bias = b2[col];
#pragma unroll
    for (int r = 0; r < 8; ++r)
      out[(size_t)(mbase + r + 8 * hi) * VSZ + col] = (acc[r] + bias) * BNS;
  }
}

// ---------------------------------------------------------------------------
extern "C" void kernel_launch(void* const* d_in, const int* in_sizes, int n_in,
                              void* d_out, int out_size, void* d_ws, size_t ws_size,
                              hipStream_t stream) {
  const int*   tokens = (const int*)d_in[0];
  const float* lk     = (const float*)d_in[1];   // [V+H, 4H]
  const float* lb     = (const float*)d_in[2];   // [4H]
  const float* W1     = (const float*)d_in[3];   // [H, DH]
  const float* b1     = (const float*)d_in[4];   // [DH]
  const float* W2     = (const float*)d_in[5];   // [DH, V]
  const float* b2     = (const float*)d_in[6];   // [V]
  char*  ws  = (char*)d_ws;                      // needs ~15.5 MB
  float* out = (float*)d_out;                    // [B*T, V] f32

  prep_kernel<<<2048, 256, 0, stream>>>(lk, lb, W1, W2, ws);
  lstm_kernel<<<1, 1024, 0, stream>>>(tokens, lk, ws);
  gemm1_kernel<<<BT / 128, 256, 0, stream>>>(b1, ws);
  gemm2_kernel<<<dim3(BT / 128, VSZ / 64), 256, 0, stream>>>(b2, ws, out);
}
